// BiDirectionalMinGRU_82471962017932
// MI455X (gfx1250) — matrix operand order repaired and checked by reading, not compiled, not run on hardware
//
#include <hip/hip_runtime.h>
#include <math.h>

// ---------------- problem constants ----------------
#define Bn    16
#define Ln    4096
#define Hn    512
#define NTn   8
#define INn   10          // 2 + NT
#define OUTn  1032        // 2H + NT
#define HHn   256
#define OUTPn 1056        // OUTn padded to multiple of 32
#define BLn   (Bn * Ln)   // 65536 tokens
#define EPSn  1e-5f
#define NCH   8           // scan chunks per sequence
#define CLEN  (Ln / NCH)  // 512

// ---------------- WMMA vector types ----------------
typedef __attribute__((ext_vector_type(16))) __bf16 v16bf;
typedef __attribute__((ext_vector_type(8)))  __bf16 v8bf;
typedef __attribute__((ext_vector_type(8)))  float  v8f;

__device__ __forceinline__ float  bf2f(__bf16 v) { return (float)v; }
__device__ __forceinline__ __bf16 f2bf(float v)  { return (__bf16)v; }

// A fragment: 16x32 bf16, lane = 16*half + m. Lane holds row m = lane&15,
// K = k0 + half*16 + e (one contiguous 32B chunk).
__device__ __forceinline__ v16bf load_a_frag(const __bf16* __restrict__ rowptr,
                                             int k0, int half) {
  return *reinterpret_cast<const v16bf*>(rowptr + k0 + half * 16);
}

// B fragment: 32x16 bf16, lane holds column n = lane&15,
// K = k0 + half*16 + e (one contiguous 32B chunk of the weight row).
__device__ __forceinline__ v16bf load_b_frag(const __bf16* __restrict__ wrow,
                                             int k0, int half) {
  return *reinterpret_cast<const v16bf*>(wrow + k0 + half * 16);
}

#define WMMA_BF16(af, bf, acc) \
  __builtin_amdgcn_wmma_f32_16x16x32_bf16(false, (af), false, (bf), (short)0, (acc), false, false)

// ---------------- K0: weight conversion / padding ----------------
__global__ __launch_bounds__(256) void k_convert(
    const float* __restrict__ fz, const float* __restrict__ fh,
    const float* __restrict__ bz, const float* __restrict__ bh,
    const float* __restrict__ w1,
    __bf16* __restrict__ wzf, __bf16* __restrict__ whf,
    __bf16* __restrict__ wzb, __bf16* __restrict__ whb,
    __bf16* __restrict__ w1p) {
  int i = blockIdx.x * 256 + threadIdx.x;
  if (i < Hn * Hn) {
    wzf[i] = f2bf(fz[i]); whf[i] = f2bf(fh[i]);
    wzb[i] = f2bf(bz[i]); whb[i] = f2bf(bh[i]);
  }
  if (i < HHn * OUTPn) {
    int n = i / OUTPn, k = i % OUTPn;
    w1p[i] = (k < OUTn) ? f2bf(w1[(size_t)n * OUTn + k]) : f2bf(0.f);
  }
}

// ---------------- K1: time embedding + concat ----------------
__global__ __launch_bounds__(256) void k_te(
    const float* __restrict__ x, const float* __restrict__ t,
    const float* __restrict__ w1, const float* __restrict__ b1,
    const float* __restrict__ w2, const float* __restrict__ b2,
    float* __restrict__ te, float* __restrict__ xc) {
  size_t i = (size_t)blockIdx.x * 256 + threadIdx.x;
  if (i >= (size_t)BLn) return;
  int b = (int)(i / Ln);
  float ts = t[i] - t[(size_t)b * Ln];
  float h1[NTn];
#pragma unroll
  for (int j = 0; j < NTn; ++j) h1[j] = fmaxf(ts * w1[j] + b1[j], 0.f);
  xc[i * INn + 0] = x[i * 2 + 0];
  xc[i * INn + 1] = x[i * 2 + 1];
#pragma unroll
  for (int j = 0; j < NTn; ++j) {
    float s = b2[j];
#pragma unroll
    for (int k = 0; k < NTn; ++k) s += w2[j * NTn + k] * h1[k];
    te[i * NTn + j] = s;
    xc[i * INn + 2 + j] = s;
  }
}

// ---------------- K2: input projection (K=10, VALU) ----------------
__global__ __launch_bounds__(256) void k_proj(
    const float* __restrict__ xc,
    const float* __restrict__ fw, const float* __restrict__ fb,
    const float* __restrict__ bw, const float* __restrict__ bbias,
    __bf16* __restrict__ xp) {
  size_t i = (size_t)blockIdx.x * 256 + threadIdx.x;  // < 2*BL*H
  int c = (int)(i & (Hn - 1));
  int dir = (int)((i >> 9) & 1);
  size_t tok = i >> 10;
  int b = (int)(tok / Ln), l = (int)(tok % Ln);
  int sl = dir ? (Ln - 1 - l) : l;  // backward direction reads reversed xc
  const float* w = (dir ? bw : fw) + (size_t)c * INn;
  float s = (dir ? bbias : fb)[c];
  const float* xr = xc + ((size_t)b * Ln + sl) * INn;
#pragma unroll
  for (int k = 0; k < INn; ++k) s += w[k] * xr[k];
  xp[((size_t)dir * BLn + tok) * Hn + c] = f2bf(s);
}

// ---------------- K3: WMMA gate GEMM + fused sigmoid epilogue ----------
// grid = (H/64, BL/64, 2dirs). 8 waves: waves 0-3 -> z, waves 4-7 -> h~.
// Each wave computes a 32x32 patch (2 M-tiles x 2 N-tiles = 4 accumulators):
// per k-step 2 A-frags + 2 B-frags feed 4 WMMAs (32 B/lane/WMMA).
__global__ __launch_bounds__(256) void k_gates(
    const __bf16* __restrict__ xp,
    const __bf16* __restrict__ wzf, const __bf16* __restrict__ whf,
    const __bf16* __restrict__ wzb, const __bf16* __restrict__ whb,
    const float* __restrict__ fz_b, const float* __restrict__ fh_b,
    const float* __restrict__ bz_b, const float* __restrict__ bh_b,
    float* __restrict__ a_out, float* __restrict__ b_out) {
  __shared__ float sbuf[2][64][64];  // [gate][row][col], 32 KB
  const int wave = threadIdx.x >> 5, lane = threadIdx.x & 31;
  const int half = lane >> 4, ln = lane & 15;
  const int which = wave >> 2;        // 0 = z, 1 = h~
  const int msub = (wave >> 1) & 1;   // M 32-half within 64
  const int nsub = wave & 1;          // N 32-half within 64
  const int dir = blockIdx.z;
  const size_t m0 = (size_t)blockIdx.y * 64;
  const int nb = blockIdx.x * 64 + nsub * 32;

  const __bf16* A0 = xp + ((size_t)dir * BLn + m0 + msub * 32 + ln) * Hn;
  const __bf16* A1 = A0 + (size_t)16 * Hn;
  const __bf16* W = which ? (dir ? whb : whf) : (dir ? wzb : wzf);
  const __bf16* W0 = W + (size_t)(nb + ln) * Hn;
  const __bf16* W1 = W0 + (size_t)16 * Hn;

  v8f acc00 = {}, acc01 = {}, acc10 = {}, acc11 = {};
#pragma unroll 4
  for (int k0 = 0; k0 < Hn; k0 += 32) {
    v16bf a0 = load_a_frag(A0, k0, half);
    v16bf a1 = load_a_frag(A1, k0, half);
    v16bf b0 = load_b_frag(W0, k0, half);
    v16bf b1 = load_b_frag(W1, k0, half);
    acc00 = WMMA_BF16(a0, b0, acc00);
    acc01 = WMMA_BF16(a0, b1, acc01);
    acc10 = WMMA_BF16(a1, b0, acc10);
    acc11 = WMMA_BF16(a1, b1, acc11);
  }
#pragma unroll
  for (int r = 0; r < 8; ++r) {
    int rr0 = msub * 32 + r + 8 * half;       // rows of tile mi=0
    int rr1 = rr0 + 16;                        // rows of tile mi=1
    int cc0 = nsub * 32 + ln;                  // cols of tile ni=0
    int cc1 = cc0 + 16;                        // cols of tile ni=1
    sbuf[which][rr0][cc0] = acc00[r];
    sbuf[which][rr0][cc1] = acc01[r];
    sbuf[which][rr1][cc0] = acc10[r];
    sbuf[which][rr1][cc1] = acc11[r];
  }
  __syncthreads();

  const float* zb = dir ? bz_b : fz_b;
  const float* hb = dir ? bh_b : fh_b;
#pragma unroll
  for (int tsub = 0; tsub < 16; ++tsub) {
    int j = threadIdx.x + tsub * 256;          // 64*64 = 4096 outputs
    int row = j >> 6, col = j & 63;
    int c = blockIdx.x * 64 + col;
    size_t idx = ((size_t)dir * BLn + m0 + row) * Hn + c;
    float z = 1.f / (1.f + __expf(-(sbuf[0][row][col] + zb[c])));
    float ht = sbuf[1][row][col] + hb[c];
    a_out[idx] = 1.f - z;
    b_out[idx] = z * ht;
  }
}

// ---------------- K4a/b/c: chunked linear-recurrence scan ----------------
// h[l] = a[l]*h[l-1] + b[l]; segment composition (P,S): h_out = P*h_in + S
__global__ __launch_bounds__(256) void k_scan1(
    const float* __restrict__ a, const float* __restrict__ bb,
    float* __restrict__ P, float* __restrict__ S) {
  size_t i = (size_t)blockIdx.x * 256 + threadIdx.x;  // c fastest
  int c = (int)(i % Hn);
  int ch = (int)((i / Hn) % NCH);
  int b = (int)((i / ((size_t)Hn * NCH)) % Bn);
  int dir = (int)(i / ((size_t)Hn * NCH * Bn));
  size_t base = (((size_t)dir * Bn + b) * Ln + (size_t)ch * CLEN) * Hn + c;
  float p = 1.f, s = 0.f;
  for (int j = 0; j < CLEN; ++j) {
    float av = a[base + (size_t)j * Hn];
    float bv = bb[base + (size_t)j * Hn];
    s = av * s + bv;
    p *= av;
  }
  P[i] = p; S[i] = s;
}

__global__ __launch_bounds__(256) void k_scan2(
    const float* __restrict__ P, const float* __restrict__ S,
    float* __restrict__ Hst) {
  size_t i = (size_t)blockIdx.x * 256 + threadIdx.x;  // < 2*B*H, c fastest
  int c = (int)(i % Hn);
  int b = (int)((i / Hn) % Bn);
  int dir = (int)(i / ((size_t)Hn * Bn));
  float h = 0.f;
  for (int ch = 0; ch < NCH; ++ch) {
    size_t idx = (size_t)c + (size_t)Hn * ((size_t)ch + NCH * ((size_t)b + (size_t)Bn * dir));
    Hst[idx] = h;
    h = P[idx] * h + S[idx];
  }
}

__global__ __launch_bounds__(256) void k_scan3(
    const float* __restrict__ a, const float* __restrict__ bb,
    const float* __restrict__ Hst, __bf16* __restrict__ hs) {
  size_t i = (size_t)blockIdx.x * 256 + threadIdx.x;
  int c = (int)(i % Hn);
  int ch = (int)((i / Hn) % NCH);
  int b = (int)((i / ((size_t)Hn * NCH)) % Bn);
  int dir = (int)(i / ((size_t)Hn * NCH * Bn));
  size_t base = (((size_t)dir * Bn + b) * Ln + (size_t)ch * CLEN) * Hn + c;
  size_t row0 = ((size_t)dir * Bn + b) * Ln;  // token row base for this (dir,b)
  float h = Hst[i];
  if (ch == 0) {  // zero-fill the shifted boundary position
    size_t zl = dir ? (size_t)(Ln - 1) : 0;
    hs[(row0 + zl) * Hn + c] = f2bf(0.f);
  }
  for (int j = 0; j < CLEN; ++j) {
    int gl = ch * CLEN + j;  // scan index (reversed l for dir=1)
    float av = a[base + (size_t)j * Hn];
    float bv = bb[base + (size_t)j * Hn];
    h = av * h + bv;
    if (dir == 0) {
      if (gl < Ln - 1) hs[(row0 + gl + 1) * Hn + c] = f2bf(h);         // hf[l-1]
    } else {
      if (gl <= Ln - 2) hs[(row0 + (Ln - 2 - gl)) * Hn + c] = f2bf(h); // hb[l+1]
    }
  }
}

// ---------------- K5: LayerNorm over 1032 ch, one wave per token -----------
__global__ __launch_bounds__(256) void k_ln(
    const __bf16* __restrict__ hs, const float* __restrict__ te,
    const float* __restrict__ g, const float* __restrict__ beta,
    const float* __restrict__ tscale, __bf16* __restrict__ hn) {
  const int wave = threadIdx.x >> 5, lane = threadIdx.x & 31;
  const size_t tok = (size_t)blockIdx.x * 8 + wave;
  float sum = 0.f, sq = 0.f;
  for (int c = lane; c < OUTn; c += 32) {
    float v;
    if (c < Hn)            v = bf2f(hs[tok * Hn + c]);
    else if (c < 2 * Hn)   v = bf2f(hs[((size_t)BLn + tok) * Hn + (c - Hn)]);
    else                   v = te[tok * NTn + (c - 2 * Hn)];
    sum += v; sq += v * v;
  }
#pragma unroll
  for (int o = 16; o > 0; o >>= 1) {
    sum += __shfl_xor(sum, o, 32);
    sq  += __shfl_xor(sq, o, 32);
  }
  float mu = sum * (1.f / OUTn);
  float var = sq * (1.f / OUTn) - mu * mu;
  float rs = rsqrtf(var + EPSn);
  float tsc = tscale[0];
  for (int c = lane; c < OUTn; c += 32) {
    float v;
    if (c < Hn)            v = bf2f(hs[tok * Hn + c]);
    else if (c < 2 * Hn)   v = bf2f(hs[((size_t)BLn + tok) * Hn + (c - Hn)]);
    else                   v = te[tok * NTn + (c - 2 * Hn)];
    float o = (v - mu) * rs * g[c] + beta[c];
    if (c >= 2 * Hn) o *= tsc;
    hn[tok * OUTPn + c] = f2bf(o);
  }
  for (int c = OUTn + lane; c < OUTPn; c += 32) hn[tok * OUTPn + c] = f2bf(0.f);
}

// ---------------- K6: WMMA head GEMM + GELU + dot(gh_w2) -------------------
// grid = BL/64; 8 waves: mt = wave>>2 (2 M-halves of 32), ng = wave&3 (4x64 N)
// Each wave: 32x64 patch = 8 accumulators; 2 A-frags + 4 B-frags per 8 WMMAs.
__global__ __launch_bounds__(256) void k_head(
    const __bf16* __restrict__ hn, const __bf16* __restrict__ w1p,
    const float* __restrict__ bias1, const float* __restrict__ w2,
    const float* __restrict__ bias2, float* __restrict__ out) {
  __shared__ float rowsum[64];
  const int wave = threadIdx.x >> 5, lane = threadIdx.x & 31;
  const int half = lane >> 4, ln = lane & 15;
  const int mt = wave >> 2, ng = wave & 3;
  const size_t m0 = (size_t)blockIdx.x * 64 + (size_t)mt * 32;

  if (threadIdx.x < 64) rowsum[threadIdx.x] = 0.f;
  __syncthreads();

  const __bf16* A0 = hn + (m0 + ln) * OUTPn;
  const __bf16* A1 = A0 + (size_t)16 * OUTPn;
  const __bf16* wr0 = w1p + (size_t)(ng * 64 + 0 + ln) * OUTPn;
  const __bf16* wr1 = w1p + (size_t)(ng * 64 + 16 + ln) * OUTPn;
  const __bf16* wr2 = w1p + (size_t)(ng * 64 + 32 + ln) * OUTPn;
  const __bf16* wr3 = w1p + (size_t)(ng * 64 + 48 + ln) * OUTPn;

  v8f acc[2][4] = {{{}, {}, {}, {}}, {{}, {}, {}, {}}};
  for (int k0 = 0; k0 < OUTPn; k0 += 32) {
    v16bf a0 = load_a_frag(A0, k0, half);
    v16bf a1 = load_a_frag(A1, k0, half);
    v16bf b0 = load_b_frag(wr0, k0, half);
    v16bf b1 = load_b_frag(wr1, k0, half);
    v16bf b2 = load_b_frag(wr2, k0, half);
    v16bf b3 = load_b_frag(wr3, k0, half);
    acc[0][0] = WMMA_BF16(a0, b0, acc[0][0]);
    acc[0][1] = WMMA_BF16(a0, b1, acc[0][1]);
    acc[0][2] = WMMA_BF16(a0, b2, acc[0][2]);
    acc[0][3] = WMMA_BF16(a0, b3, acc[0][3]);
    acc[1][0] = WMMA_BF16(a1, b0, acc[1][0]);
    acc[1][1] = WMMA_BF16(a1, b1, acc[1][1]);
    acc[1][2] = WMMA_BF16(a1, b2, acc[1][2]);
    acc[1][3] = WMMA_BF16(a1, b3, acc[1][3]);
  }

  float part[2][8];
#pragma unroll
  for (int mi = 0; mi < 2; ++mi)
#pragma unroll
    for (int r = 0; r < 8; ++r) part[mi][r] = 0.f;
#pragma unroll
  for (int j = 0; j < 4; ++j) {
    int n = ng * 64 + j * 16 + ln;
    float b1v = bias1[n], w2v = w2[n];
#pragma unroll
    for (int mi = 0; mi < 2; ++mi) {
#pragma unroll
      for (int r = 0; r < 8; ++r) {
        float v = acc[mi][j][r] + b1v;
        float gl = 0.5f * v * (1.f + erff(v * 0.70710678118654752f));  // exact GELU
        part[mi][r] += gl * w2v;
      }
    }
  }
#pragma unroll
  for (int mi = 0; mi < 2; ++mi)
#pragma unroll
    for (int r = 0; r < 8; ++r)
      atomicAdd(&rowsum[mt * 32 + mi * 16 + r + 8 * half], part[mi][r]);  // ds_add_f32
  __syncthreads();
  if (threadIdx.x < 64)
    out[(size_t)blockIdx.x * 64 + threadIdx.x] = rowsum[threadIdx.x] + bias2[0];
}

// ---------------- host launcher ----------------
extern "C" void kernel_launch(void* const* d_in, const int* in_sizes, int n_in,
                              void* d_out, int out_size, void* d_ws,
                              size_t ws_size, hipStream_t stream) {
  const float* x      = (const float*)d_in[0];
  const float* t      = (const float*)d_in[1];
  const float* te_w1  = (const float*)d_in[2];
  const float* te_b1  = (const float*)d_in[3];
  const float* te_w2  = (const float*)d_in[4];
  const float* te_b2  = (const float*)d_in[5];
  const float* fproj_w = (const float*)d_in[6];
  const float* fproj_b = (const float*)d_in[7];
  const float* bproj_w = (const float*)d_in[8];
  const float* bproj_b = (const float*)d_in[9];
  const float* fz_w = (const float*)d_in[10];
  const float* fz_b = (const float*)d_in[11];
  const float* fh_w = (const float*)d_in[12];
  const float* fh_b = (const float*)d_in[13];
  const float* bz_w = (const float*)d_in[14];
  const float* bz_b = (const float*)d_in[15];
  const float* bh_w = (const float*)d_in[16];
  const float* bh_b = (const float*)d_in[17];
  const float* ln_g = (const float*)d_in[18];
  const float* ln_b = (const float*)d_in[19];
  const float* tsc  = (const float*)d_in[20];
  const float* gh_w1 = (const float*)d_in[21];
  const float* gh_b1 = (const float*)d_in[22];
  const float* gh_w2 = (const float*)d_in[23];
  const float* gh_b2 = (const float*)d_in[24];
  float* out = (float*)d_out;

  // workspace layout (peak ~675 MB; hshift reuses xp, hn reuses a)
  char* ws = (char*)d_ws;
  size_t off = 0;
  auto take = [&](size_t bytes) -> char* {
    char* p = ws + off;
    off += (bytes + 255) & ~(size_t)255;
    return p;
  };
  __bf16* wzf = (__bf16*)take((size_t)Hn * Hn * 2);
  __bf16* whf = (__bf16*)take((size_t)Hn * Hn * 2);
  __bf16* wzb = (__bf16*)take((size_t)Hn * Hn * 2);
  __bf16* whb = (__bf16*)take((size_t)Hn * Hn * 2);
  __bf16* w1p = (__bf16*)take((size_t)HHn * OUTPn * 2);
  float*  te  = (float*)take((size_t)BLn * NTn * 4);
  float*  xcb = (float*)take((size_t)BLn * INn * 4);
  __bf16* xp  = (__bf16*)take((size_t)2 * BLn * Hn * 2);   // later reused: hshift
  float*  abuf = (float*)take((size_t)2 * BLn * Hn * 4);   // later reused: hn
  float*  bbuf = (float*)take((size_t)2 * BLn * Hn * 4);
  float*  Pb  = (float*)take((size_t)2 * Bn * NCH * Hn * 4);
  float*  Sb  = (float*)take((size_t)2 * Bn * NCH * Hn * 4);
  float*  Hst = (float*)take((size_t)2 * Bn * NCH * Hn * 4);
  __bf16* hs = xp;             // xp dead after k_gates
  __bf16* hn = (__bf16*)abuf;  // a/b dead after k_scan3

  k_convert<<<(HHn * OUTPn + 255) / 256, 256, 0, stream>>>(
      fz_w, fh_w, bz_w, bh_w, gh_w1, wzf, whf, wzb, whb, w1p);
  k_te<<<BLn / 256, 256, 0, stream>>>(x, t, te_w1, te_b1, te_w2, te_b2, te, xcb);
  k_proj<<<(unsigned)(((size_t)2 * BLn * Hn) / 256), 256, 0, stream>>>(
      xcb, fproj_w, fproj_b, bproj_w, bproj_b, xp);
  k_gates<<<dim3(Hn / 64, BLn / 64, 2), 256, 0, stream>>>(
      xp, wzf, whf, wzb, whb, fz_b, fh_b, bz_b, bh_b, abuf, bbuf);
  k_scan1<<<(2 * Bn * NCH * Hn) / 256, 256, 0, stream>>>(abuf, bbuf, Pb, Sb);
  k_scan2<<<(2 * Bn * Hn) / 256, 256, 0, stream>>>(Pb, Sb, Hst);
  k_scan3<<<(2 * Bn * NCH * Hn) / 256, 256, 0, stream>>>(abuf, bbuf, Hst, hs);
  k_ln<<<BLn / 8, 256, 0, stream>>>(hs, te, ln_g, ln_b, tsc, hn);
  k_head<<<BLn / 64, 256, 0, stream>>>(hn, w1p, gh_b1, gh_w2, gh_b2, out);
}
